// STGNN_17145509446140
// MI455X (gfx1250) — matrix operand order repaired and checked by reading, not compiled, not run on hardware
//
#include <hip/hip_runtime.h>

#define HID 128

typedef __attribute__((ext_vector_type(16))) __bf16 v16bf;
typedef __attribute__((ext_vector_type(8)))  float  v8f;

__device__ __forceinline__ __bf16 f2bf(float x) { return (__bf16)x; }

// -------------------- degree / symmetric norm --------------------
__global__ void k_zero(float* __restrict__ p, int n) {
  int i = blockIdx.x * blockDim.x + threadIdx.x;
  if (i < n) p[i] = 0.0f;
}

__global__ void k_deg(const int* __restrict__ dst, float* __restrict__ deg, int E) {
  int e = blockIdx.x * blockDim.x + threadIdx.x;
  if (e < E) unsafeAtomicAdd(&deg[dst[e]], 1.0f);
}

__global__ void k_dinv(float* __restrict__ d, int n) {
  int i = blockIdx.x * blockDim.x + threadIdx.x;
  if (i < n) {
    float v = d[i] + 1.0f;          // self-loop
    d[i] = (v > 0.0f) ? rsqrtf(v) : 0.0f;
  }
}

// -------------------- layer-1 GEMM (K = 5, scalar) --------------------
__global__ void k_gemm_in(const float* __restrict__ x, const float* __restrict__ W,
                          float* __restrict__ out, int nel) {   // nel = N*128
  int i = blockIdx.x * blockDim.x + threadIdx.x;
  if (i >= nel) return;
  int node = i >> 7;
  int col  = i & 127;
  const float* xr = x + (size_t)node * 5;
  float s = xr[0] * W[          col] + xr[1] * W[128 + col] + xr[2] * W[256 + col]
          + xr[3] * W[384 + col] + xr[4] * W[512 + col];
  out[i] = s;
}

// -------------------- aggregation --------------------
// self-loop term initializes the output buffer: out = h * dinv^2
__global__ void k_agg_init(const float* __restrict__ h, const float* __restrict__ dinv,
                           float* __restrict__ out, int nq) {   // nq = N*32 (float4 units)
  int i = blockIdx.x * blockDim.x + threadIdx.x;
  if (i >= nq) return;
  int node = i >> 5;
  float di = dinv[node];
  float w = di * di;
  float4 v = reinterpret_cast<const float4*>(h)[i];
  v.x *= w; v.y *= w; v.z *= w; v.w *= w;
  reinterpret_cast<float4*>(out)[i] = v;
}

// one wave per edge: lane l handles floats [4l, 4l+3] of the 128-wide row
__global__ void k_agg_edges(const float* __restrict__ h, const int* __restrict__ src,
                            const int* __restrict__ dst, const float* __restrict__ dinv,
                            float* __restrict__ out, int E) {
  int gid  = blockIdx.x * blockDim.x + threadIdx.x;
  int e    = gid >> 5;
  int lane = threadIdx.x & 31;
  if (e >= E) return;
  int s = src[e], d = dst[e];
  float nrm = dinv[s] * dinv[d];
  float4 v = reinterpret_cast<const float4*>(h + (size_t)s * HID)[lane];
  float* o = out + (size_t)d * HID + lane * 4;
  unsafeAtomicAdd(o + 0, v.x * nrm);
  unsafeAtomicAdd(o + 1, v.y * nrm);
  unsafeAtomicAdd(o + 2, v.z * nrm);
  unsafeAtomicAdd(o + 3, v.w * nrm);
}

__global__ void k_bias_relu(float* __restrict__ h, const float* __restrict__ b, int nq) {
  int i = blockIdx.x * blockDim.x + threadIdx.x;
  if (i >= nq) return;
  int c4 = (i & 31) * 4;
  float4 v = reinterpret_cast<float4*>(h)[i];
  v.x = fmaxf(v.x + b[c4 + 0], 0.0f);
  v.y = fmaxf(v.y + b[c4 + 1], 0.0f);
  v.z = fmaxf(v.z + b[c4 + 2], 0.0f);
  v.w = fmaxf(v.w + b[c4 + 3], 0.0f);
  reinterpret_cast<float4*>(h)[i] = v;
}

// -------------------- 128x128 GEMM via WMMA bf16 --------------------
// C[tile*16 .. +16, :] = A[tile*16 .. +16, :] @ W (128x128), f32 in/out,
// bf16 WMMA with f32 accumulation. 8 waves/block, wave w owns columns w*16..+15.
__global__ void k_gemm_wmma(const float* __restrict__ A, const float* __restrict__ W,
                            float* __restrict__ C, int ntiles) {
  const int lane    = threadIdx.x & 31;
  const int wave    = threadIdx.x >> 5;      // 0..7
  const int coltile = wave * 16;
  const int col     = lane & 15;
  const int hi      = lane >> 4;             // 0: lanes 0-15, 1: lanes 16-31

  // B fragments held in registers across the whole tile loop.
  // B 32x16 bf16, contiguous k order used by this code: lanes 0-15 -> N=lane, K=0..15; lanes 16-31 -> K=16..31.
  v16bf bfrag[4];
#pragma unroll
  for (int ks = 0; ks < 4; ++ks) {
    int kb = ks * 32 + hi * 16;
    const float* wp = W + (size_t)kb * HID + coltile + col;
    v16bf b;
#pragma unroll
    for (int eidx = 0; eidx < 16; ++eidx)
      b[eidx] = f2bf(wp[(size_t)eidx * HID]);
    bfrag[ks] = b;
  }

  const int row   = lane & 15;
  const int khalf = hi * 16;                 // A K offset per half (contiguous k order, matches B)

  for (int tile = blockIdx.x; tile < ntiles; tile += gridDim.x) {
    const float* ab = A + (size_t)tile * 16 * HID + (size_t)row * HID + khalf;
    v8f acc = {0.f, 0.f, 0.f, 0.f, 0.f, 0.f, 0.f, 0.f};
#pragma unroll
    for (int ks = 0; ks < 4; ++ks) {
      const float* ap = ab + ks * 32;
      float4 p0 = *reinterpret_cast<const float4*>(ap + 0);
      float4 p1 = *reinterpret_cast<const float4*>(ap + 4);
      float4 p2 = *reinterpret_cast<const float4*>(ap + 8);
      float4 p3 = *reinterpret_cast<const float4*>(ap + 12);
      v16bf a;
      a[0]  = f2bf(p0.x); a[1]  = f2bf(p0.y); a[2]  = f2bf(p0.z); a[3]  = f2bf(p0.w);
      a[4]  = f2bf(p1.x); a[5]  = f2bf(p1.y); a[6]  = f2bf(p1.z); a[7]  = f2bf(p1.w);
      a[8]  = f2bf(p2.x); a[9]  = f2bf(p2.y); a[10] = f2bf(p2.z); a[11] = f2bf(p2.w);
      a[12] = f2bf(p3.x); a[13] = f2bf(p3.y); a[14] = f2bf(p3.z); a[15] = f2bf(p3.w);
      acc = __builtin_amdgcn_wmma_f32_16x16x32_bf16(false, a, false, bfrag[ks],
                                                    (short)0, acc, false, false);
    }
    // C 16x16 f32 layout: VGPR r -> M = r + hi*8, N = col
    float* cb = C + (size_t)tile * 16 * HID + (size_t)(hi * 8) * HID + coltile + col;
#pragma unroll
    for (int r = 0; r < 8; ++r)
      cb[(size_t)r * HID] = acc[r];
  }
}

// scalar tail for N % 16 (not hit for N=100000, kept for generality)
__global__ void k_gemm_tail(const float* __restrict__ A, const float* __restrict__ W,
                            float* __restrict__ C, int startNode, int nel) {
  int i = blockIdx.x * blockDim.x + threadIdx.x;
  if (i >= nel) return;
  int node = startNode + (i >> 7);
  int col  = i & 127;
  const float* ar = A + (size_t)node * HID;
  float s = 0.0f;
  for (int k = 0; k < HID; ++k) s += ar[k] * W[(size_t)k * HID + col];
  C[(size_t)node * HID + col] = s;
}

// -------------------- final FC (128 -> 1), wave reduction --------------------
__global__ void k_fc(const float* __restrict__ h, const float* __restrict__ Wfc,
                     const float* __restrict__ bfc, float* __restrict__ out, int N) {
  int gid  = blockIdx.x * blockDim.x + threadIdx.x;
  int node = gid >> 5;
  int lane = threadIdx.x & 31;
  if (node >= N) return;
  float4 v = reinterpret_cast<const float4*>(h + (size_t)node * HID)[lane];
  float4 w = reinterpret_cast<const float4*>(Wfc)[lane];
  float s = v.x * w.x + v.y * w.y + v.z * w.z + v.w * w.w;
#pragma unroll
  for (int off = 16; off > 0; off >>= 1)
    s += __shfl_down(s, off, 32);
  if (lane == 0) out[node] = s + bfc[0];
}

// -------------------- host orchestration --------------------
static inline void aggregate_layer(const float* hin, float* hout, const float* dinv,
                                   const int* src, const int* dst, const float* bias,
                                   int N, int E, hipStream_t stream) {
  int nq = N * 32;
  k_agg_init<<<(nq + 255) / 256, 256, 0, stream>>>(hin, dinv, hout, nq);
  k_agg_edges<<<(E + 7) / 8, 256, 0, stream>>>(hin, src, dst, dinv, hout, E);
  k_bias_relu<<<(nq + 255) / 256, 256, 0, stream>>>(hout, bias, nq);
}

static inline void gemm128(const float* A, const float* W, float* C, int N,
                           hipStream_t stream) {
  int ntiles = N / 16;
  if (ntiles > 0) {
    int grid = ntiles < 1024 ? ntiles : 1024;
    k_gemm_wmma<<<grid, 256, 0, stream>>>(A, W, C, ntiles);
  }
  int rem = N & 15;
  if (rem) {
    int nel = rem * HID;
    k_gemm_tail<<<(nel + 255) / 256, 256, 0, stream>>>(A, W, C, N - rem, nel);
  }
}

extern "C" void kernel_launch(void* const* d_in, const int* in_sizes, int n_in,
                              void* d_out, int out_size, void* d_ws, size_t ws_size,
                              hipStream_t stream) {
  const float* x   = (const float*)d_in[0];
  const int*   ei  = (const int*)d_in[1];
  const float* W1  = (const float*)d_in[2];
  const float* b1  = (const float*)d_in[3];
  const float* W2  = (const float*)d_in[4];
  const float* b2  = (const float*)d_in[5];
  const float* W3  = (const float*)d_in[6];
  const float* b3  = (const float*)d_in[7];
  const float* Wfc = (const float*)d_in[8];
  const float* bfc = (const float*)d_in[9];
  float* out = (float*)d_out;

  const int N = in_sizes[0] / 5;      // 100000
  const int E = in_sizes[1] / 2;      // 1600000
  const int* src = ei;                // edge_index row 0
  const int* dst = ei + E;            // edge_index row 1

  // workspace: dinv[N] | hA[N*128] | hB[N*128]  (f32)
  float* base = (float*)d_ws;
  float* dinv = base;
  float* hA   = base + (size_t)((N + 511) & ~511);
  float* hB   = hA + (size_t)N * HID;

  // 1) symmetric GCN norm factors
  k_zero<<<(N + 255) / 256, 256, 0, stream>>>(dinv, N);
  k_deg<<<(E + 255) / 256, 256, 0, stream>>>(dst, dinv, E);
  k_dinv<<<(N + 255) / 256, 256, 0, stream>>>(dinv, N);

  // 2) layer 1: h = x @ W1 ; aggregate ; +b1 ; relu
  int nel = N * HID;
  k_gemm_in<<<(nel + 255) / 256, 256, 0, stream>>>(x, W1, hA, nel);
  aggregate_layer(hA, hB, dinv, src, dst, b1, N, E, stream);

  // 3) layer 2: WMMA GEMM ; aggregate ; +b2 ; relu
  gemm128(hB, W2, hA, N, stream);
  aggregate_layer(hA, hB, dinv, src, dst, b2, N, E, stream);

  // 4) layer 3
  gemm128(hB, W3, hA, N, stream);
  aggregate_layer(hA, hB, dinv, src, dst, b3, N, E, stream);

  // 5) final FC 128 -> 1
  k_fc<<<(N + 7) / 8, 256, 0, stream>>>(hB, Wfc, bfc, out, N);
}
